// EGNNLayer_62431644614834
// MI455X (gfx1250) — matrix operand order repaired and checked by reading, not compiled, not run on hardware
//
#include <hip/hip_runtime.h>
#include <hip/hip_bf16.h>
#include <math.h>

// ---------------------------------------------------------------------------
// EGNN layer for MI455X (gfx1250, wave32, WMMA + TDM).
// Stages:
//   P: qa = query@a1, ka = key_w@a2, e_att = weight_e . a3      (tiny)
//   I: zero agg / ssum / segmax
//   1: h_src = feat_src@Wn (f32 out), h_dst = feat_dst@Wn (bf16 out),
//      s_q[n] = h_dst[n].qa, s_k[n] = h_src[n].ka           [WMMA bf16]
//   2: per-edge logit = leakyrelu(s_q[dst]+s_k[src]+e_att*ew+b); segment max
//   3: ex = exp(logit - max[dst]); segment sum
//   4: agg[dst] += (ex/sum[dst]) * h_src[src]  (wave-chunked, sorted dst ->
//      one atomic pair per segment boundary per lane)
//   5: out = [h_dst | agg] @ out_w + out_b                  [WMMA bf16]
// Weight staging in stages 1/5 uses the Tensor Data Mover (tensor_load_to_lds)
// on the device pass (builtin confirmed declared for gfx1250 by the round-4
// probe); the __has_builtin guard keeps the host pass on the fallback.
// ---------------------------------------------------------------------------

typedef __attribute__((ext_vector_type(16))) __bf16 v16bf;
typedef __attribute__((ext_vector_type(8)))  __bf16 v8bf;
typedef __attribute__((ext_vector_type(8)))  float  v8f;

static constexpr int kN   = 50000;   // nodes
static constexpr int kE   = 800000;  // edges
static constexpr int kInC = 128;     // input channels
static constexpr int kD   = 64;      // output channels (d)

#if __has_builtin(__builtin_amdgcn_tensor_load_to_lds) && \
    __has_builtin(__builtin_amdgcn_s_wait_tensorcnt)
#define EGNN_USE_TDM 1
#else
#define EGNN_USE_TDM 0
#endif

#if EGNN_USE_TDM
typedef unsigned int tdm_u32x4 __attribute__((ext_vector_type(4)));
typedef int          tdm_i32x8 __attribute__((ext_vector_type(8)));
typedef int          tdm_i32x4 __attribute__((ext_vector_type(4)));

// One-shot TDM DMA: copy 8192 f32 (128x64 weight) from global to LDS.
// D# per CDNA5 ISA: group0 = control/lds/global addr,
// group1 = dims (1-D tile: tensor_dim0 = tile_dim0 = 8192 elems of 4B),
// groups 2/3 unused (<=2D tensor).
__device__ inline void tdm_copy_8192f32_to_lds(const float* gsrc,
                                               float* lds_dst) {
  unsigned long long ga = (unsigned long long)(uintptr_t)gsrc;
  unsigned lds_off = (unsigned)(uintptr_t)lds_dst;  // flat addr[31:0] = LDS byte addr
  tdm_u32x4 g0;
  g0.x = 1u;                                         // count=1 (valid user D#)
  g0.y = lds_off;                                    // lds_addr
  g0.z = (unsigned)(ga & 0xffffffffull);             // global_addr[31:0]
  g0.w = (unsigned)((ga >> 32) & 0x01ffffffull)      // global_addr[56:32]
         | (2u << 30);                               // type=2 ("image")
  tdm_i32x8 g1;
  g1[0] = 0x00020000;                  // wg_mask=0, data_size=2 -> 4 bytes
  g1[1] = (int)((8192u & 0xffffu) << 16);   // tensor_dim0 lo16 (bits 63:48)
  g1[2] = (int)(((8192u >> 16) & 0xffffu) | (1u << 16)); // dim0 hi | tensor_dim1=1
  g1[3] = (int)((8192u & 0xffffu) << 16);   // tile_dim0 = 8192 (bits 127:112)
  g1[4] = 1;                                // tile_dim1 = 1, tile_dim2 = 0
  g1[5] = 8192;                             // tensor_dim0_stride lo32
  g1[6] = 0;                                // stride0 hi16 | stride1 lo16
  g1[7] = 0;                                // stride1 hi32
  tdm_i32x4 z4 = {0, 0, 0, 0};
#if __clang_major__ >= 23
  tdm_i32x8 z8 = {0, 0, 0, 0, 0, 0, 0, 0};
  __builtin_amdgcn_tensor_load_to_lds(g0, g1, z4, z4, z8, 0);
#else
  __builtin_amdgcn_tensor_load_to_lds(g0, g1, z4, z4, 0);
#endif
}
#endif  // EGNN_USE_TDM

// ---------------------------------------------------------------------------
// Pack a 128x64 f32 weight matrix into B-fragment order (bf16) in LDS.
// Layout: wp[((kt*4 + nt)*32 + lane)*16 + j]
//   B tile (kt,nt) is W[kt*32 .. kt*32+31, nt*16 .. nt*16+15] (32x16, KxN):
//   lane 0-15  : N = lane,    K = kt*32 + j (j<8), kt*32 + 16 + (j-8) (j>=8)
//   lane 16-31 : N = lane-16, K = kt*32 + 8 + j (j<8), kt*32 + 24 + (j-8) (j>=8)
// Each lane's fragment is 16 bf16 = 32 contiguous bytes -> aligned LDS loads.
// W may point to global memory or to an LDS staging buffer (generic ptr).
// Requires blockDim.x == 256.
// ---------------------------------------------------------------------------
__device__ inline void pack_w128x64(const float* __restrict__ W,
                                    __bf16* __restrict__ wp) {
  for (int j = 0; j < 32; ++j) {
    int flat = (int)threadIdx.x + 256 * j;       // 0 .. 8191
    int j16  = flat & 15;
    int lane = (flat >> 4) & 31;
    int nt   = (flat >> 9) & 3;
    int kt   = (flat >> 11) & 3;
    int k = kt * 32 + ((lane >> 4) * 8) + (j16 & 7) + ((j16 >> 3) * 16);
    int n = nt * 16 + (lane & 15);
    wp[flat] = (__bf16)W[k * kD + n];
  }
}

// Stage the weight matrix into bf16 fragment order in LDS, using the TDM when
// available. Must be called by all 256 threads; ends with the data ready.
__device__ inline void stage_weights(const float* __restrict__ Wg,
                                     __bf16* __restrict__ wp,
                                     float* __restrict__ wraw) {
#if EGNN_USE_TDM
  if ((threadIdx.x >> 5) == 0) {        // wave 0 issues the DMA
    tdm_copy_8192f32_to_lds(Wg, wraw);
    __builtin_amdgcn_s_wait_tensorcnt(0);
  }
  __syncthreads();
  pack_w128x64(wraw, wp);               // LDS -> LDS repack (bf16)
#else
  (void)wraw;
  pack_w128x64(Wg, wp);
#endif
  __syncthreads();
}

// A fragment (16x32 bf16, MxK) from an f32 row, per ISA layout:
//   lane<16 : elems 0-7 = K kb+0..7,  elems 8-15 = K kb+16..23
//   lane>=16: elems 0-7 = K kb+8..15, elems 8-15 = K kb+24..31
__device__ inline v16bf load_a_f32(const float* __restrict__ rowp, int kb,
                                   int lane) {
  int g0 = kb + ((lane >> 4) << 3);
  const float4* p0 = (const float4*)(rowp + g0);
  const float4* p1 = (const float4*)(rowp + g0 + 16);
  float4 x0 = p0[0], x1 = p0[1];
  float4 y0 = p1[0], y1 = p1[1];
  v16bf a;
  a[0] = (__bf16)x0.x; a[1] = (__bf16)x0.y; a[2] = (__bf16)x0.z; a[3] = (__bf16)x0.w;
  a[4] = (__bf16)x1.x; a[5] = (__bf16)x1.y; a[6] = (__bf16)x1.z; a[7] = (__bf16)x1.w;
  a[8]  = (__bf16)y0.x; a[9]  = (__bf16)y0.y; a[10] = (__bf16)y0.z; a[11] = (__bf16)y0.w;
  a[12] = (__bf16)y1.x; a[13] = (__bf16)y1.y; a[14] = (__bf16)y1.z; a[15] = (__bf16)y1.w;
  return a;
}

__device__ inline v16bf load_a_bf16(const __bf16* __restrict__ rowp, int kb,
                                    int lane) {
  int g0 = kb + ((lane >> 4) << 3);
  v8bf x = *(const v8bf*)(rowp + g0);
  v8bf y = *(const v8bf*)(rowp + g0 + 16);
  v16bf a;
#pragma unroll
  for (int i = 0; i < 8; ++i) { a[i] = x[i]; a[8 + i] = y[i]; }
  return a;
}

// Order-preserving float <-> uint mapping for atomic segment max.
__device__ inline unsigned f2ord(float f) {
  unsigned u = __float_as_uint(f);
  return (u & 0x80000000u) ? ~u : (u | 0x80000000u);
}
__device__ inline float ord2f(unsigned k) {
  return (k & 0x80000000u) ? __uint_as_float(k & 0x7fffffffu)
                           : __uint_as_float(~k);
}

// ---------------------------------------------------------------------------
// Stage P: attention-vector precompute (1 block, 64 threads).
// cst[0:64]=qa, cst[64:128]=ka, cst[128]=e_att
// ---------------------------------------------------------------------------
__global__ void precompute_kernel(const float* __restrict__ query,
                                  const float* __restrict__ key_w,
                                  const float* __restrict__ weight_e,
                                  const float* __restrict__ att_w,
                                  float* __restrict__ cst) {
  int i = threadIdx.x;  // 0..63
  float q = 0.f, k = 0.f;
  for (int j = 0; j < kD; ++j) {
    q += query[i * kD + j] * att_w[j];
    k += key_w[i * kD + j] * att_w[kD + j];
  }
  cst[i] = q;
  cst[kD + i] = k;
  if (i == 0) {
    float e = 0.f;
    for (int j = 0; j < kD; ++j) e += weight_e[j] * att_w[2 * kD + j];
    cst[2 * kD] = e;
  }
}

// ---------------------------------------------------------------------------
// Stage I: init accumulators (ws is poisoned, must be cleared).
// ---------------------------------------------------------------------------
__global__ void init_kernel(float* __restrict__ agg, float* __restrict__ ssum,
                            unsigned* __restrict__ m_enc) {
  int i = blockIdx.x * blockDim.x + threadIdx.x;
  if (i < kN * kD) agg[i] = 0.f;
  if (i < kN) { ssum[i] = 0.f; m_enc[i] = 0u; }  // 0u < f2ord(-inf)
}

// ---------------------------------------------------------------------------
// Stage 1: node transform. 256 thr = 8 waves; wave w handles rows
// blockIdx.x*128 + w*16 .. +15; 4 N-tiles of 16, K loop 128/32.
// ---------------------------------------------------------------------------
__global__ __launch_bounds__(256) void node_transform_kernel(
    const float* __restrict__ feat_src, const float* __restrict__ feat_dst,
    const float* __restrict__ weight_n, const float* __restrict__ cst,
    float* __restrict__ h_src, __bf16* __restrict__ h_dst,
    float* __restrict__ s_q, float* __restrict__ s_k) {
  __shared__ __align__(32) __bf16 wp[4 * 4 * 32 * 16];  // 16 KB packed
#if EGNN_USE_TDM
  __shared__ __align__(16) float wraw[kInC * kD];       // 32 KB TDM staging
#else
  float* wraw = nullptr;
#endif
  stage_weights(weight_n, wp, wraw);

  const int lane = threadIdx.x & 31;
  const int wave = threadIdx.x >> 5;
  const int m0 = blockIdx.x * 128 + wave * 16;
  if (m0 >= kN) return;  // uniform per wave; EXEC stays all-ones for WMMA

  float qa_[4], ka_[4];
#pragma unroll
  for (int nt = 0; nt < 4; ++nt) {
    qa_[nt] = cst[nt * 16 + (lane & 15)];
    ka_[nt] = cst[kD + nt * 16 + (lane & 15)];
  }

  const int r  = m0 + (lane & 15);
  const int rc = r < kN ? r : kN - 1;  // clamp loads; stores are guarded
  const float* rs = feat_src + (size_t)rc * kInC;
  const float* rd = feat_dst + (size_t)rc * kInC;
  __builtin_prefetch(rs, 0, 0);  // global_prefetch_b8
  __builtin_prefetch(rd, 0, 0);

  v8f hs[4] = {}, hd[4] = {};
#pragma unroll
  for (int kt = 0; kt < 4; ++kt) {
    v16bf as = load_a_f32(rs, kt * 32, lane);
    v16bf ad = load_a_f32(rd, kt * 32, lane);
#pragma unroll
    for (int nt = 0; nt < 4; ++nt) {
      v16bf b = *(const v16bf*)&wp[((kt * 4 + nt) * 32 + lane) * 16];
      hs[nt] = __builtin_amdgcn_wmma_f32_16x16x32_bf16(
          false, as, false, b, (short)0, hs[nt], false, false);
      hd[nt] = __builtin_amdgcn_wmma_f32_16x16x32_bf16(
          false, ad, false, b, (short)0, hd[nt], false, false);
    }
  }

  const int colb = lane & 15;
  const int half = lane >> 4;
#pragma unroll
  for (int i = 0; i < 8; ++i) {
    int row = m0 + i + half * 8;
    bool ok = row < kN;
    float pq = 0.f, pk = 0.f;
#pragma unroll
    for (int nt = 0; nt < 4; ++nt) {
      float vs = hs[nt][i], vd = hd[nt][i];
      if (ok) {
        h_src[(size_t)row * kD + nt * 16 + colb] = vs;
        h_dst[(size_t)row * kD + nt * 16 + colb] = (__bf16)vd;
      }
      pq += vd * qa_[nt];
      pk += vs * ka_[nt];
    }
    // reduce across the 16 lanes of each half-wave
#pragma unroll
    for (int off = 8; off > 0; off >>= 1) {
      pq += __shfl_xor(pq, off, 16);
      pk += __shfl_xor(pk, off, 16);
    }
    if (ok && colb == 0) { s_q[row] = pq; s_k[row] = pk; }
  }
}

// ---------------------------------------------------------------------------
// Stage 2: edge logits + segment max (ordered-uint atomicMax).
// ---------------------------------------------------------------------------
__global__ void edge_logits_kernel(const float* __restrict__ ew,
                                   const int* __restrict__ src,
                                   const int* __restrict__ dst,
                                   const float* __restrict__ s_q,
                                   const float* __restrict__ s_k,
                                   const float* __restrict__ cst,
                                   const float* __restrict__ att_b,
                                   float* __restrict__ logit,
                                   unsigned* __restrict__ m_enc) {
  const float eatt = cst[2 * kD];
  const float bia = att_b[0];
  for (int e = blockIdx.x * blockDim.x + threadIdx.x; e < kE;
       e += gridDim.x * blockDim.x) {
    int d = dst[e];
    float l = s_q[d] + s_k[src[e]] + eatt * ew[e] + bia;
    l = l > 0.f ? l : 0.2f * l;  // leaky_relu(0.2)
    logit[e] = l;
    atomicMax(&m_enc[d], f2ord(l));
  }
}

// ---------------------------------------------------------------------------
// Stage 3: ex = exp(logit - max); segment sum. (ex overwrites logit buffer.)
// ---------------------------------------------------------------------------
__global__ void edge_exp_kernel(const int* __restrict__ dst,
                                const unsigned* __restrict__ m_enc,
                                float* __restrict__ logit_ex,
                                float* __restrict__ ssum) {
  for (int e = blockIdx.x * blockDim.x + threadIdx.x; e < kE;
       e += gridDim.x * blockDim.x) {
    int d = dst[e];
    float x = __expf(logit_ex[e] - ord2f(m_enc[d]));
    logit_ex[e] = x;
    atomicAdd(&ssum[d], x);
  }
}

// ---------------------------------------------------------------------------
// Stage 4: segmented scatter. dst_idx is sorted, so each wave walks a
// contiguous edge chunk keeping a running accumulator per lane (2 channels),
// flushing with atomicAdd only at segment boundaries / chunk edges.
// ---------------------------------------------------------------------------
__global__ __launch_bounds__(256) void edge_scatter_kernel(
    const int* __restrict__ src, const int* __restrict__ dst,
    const float* __restrict__ ex, const float* __restrict__ ssum,
    const float* __restrict__ h_src, float* __restrict__ agg) {
  const int gt = blockIdx.x * blockDim.x + threadIdx.x;
  const int wave = gt >> 5;
  const int lane = gt & 31;
  const int nwaves = (gridDim.x * blockDim.x) >> 5;
  const int chunk = (kE + nwaves - 1) / nwaves;
  const int e0 = wave * chunk;
  const int e1 = min(kE, e0 + chunk);
  if (e0 >= kE) return;

  float a0 = 0.f, a1 = 0.f;
  int cur = -1;
  for (int e = e0; e < e1; ++e) {
    int d = dst[e];
    if (d != cur) {
      if (cur >= 0) {
        atomicAdd(&agg[(size_t)cur * kD + 2 * lane + 0], a0);
        atomicAdd(&agg[(size_t)cur * kD + 2 * lane + 1], a1);
      }
      cur = d;
      a0 = 0.f;
      a1 = 0.f;
    }
    float al = ex[e] / (ssum[d] + 1e-16f);
    float2 h = ((const float2*)(h_src + (size_t)src[e] * kD))[lane];
    a0 += al * h.x;
    a1 += al * h.y;
  }
  if (cur >= 0) {
    atomicAdd(&agg[(size_t)cur * kD + 2 * lane + 0], a0);
    atomicAdd(&agg[(size_t)cur * kD + 2 * lane + 1], a1);
  }
}

// ---------------------------------------------------------------------------
// Stage 5: out = [h_dst | agg] @ out_w + out_b.  Same WMMA tiling as stage 1;
// A's first 64 K-columns come from h_dst (bf16), last 64 from agg (f32->bf16).
// ---------------------------------------------------------------------------
__global__ __launch_bounds__(256) void out_gemm_kernel(
    const __bf16* __restrict__ h_dst, const float* __restrict__ agg,
    const float* __restrict__ out_w, const float* __restrict__ out_b,
    float* __restrict__ out) {
  __shared__ __align__(32) __bf16 wp[4 * 4 * 32 * 16];
#if EGNN_USE_TDM
  __shared__ __align__(16) float wraw[kInC * kD];
#else
  float* wraw = nullptr;
#endif
  stage_weights(out_w, wp, wraw);

  const int lane = threadIdx.x & 31;
  const int wave = threadIdx.x >> 5;
  const int m0 = blockIdx.x * 128 + wave * 16;
  if (m0 >= kN) return;

  float ob[4];
#pragma unroll
  for (int nt = 0; nt < 4; ++nt) ob[nt] = out_b[nt * 16 + (lane & 15)];

  const int r  = m0 + (lane & 15);
  const int rc = r < kN ? r : kN - 1;
  const __bf16* hdp = h_dst + (size_t)rc * kD;
  const float*  agp = agg + (size_t)rc * kD;
  __builtin_prefetch(hdp, 0, 0);
  __builtin_prefetch(agp, 0, 0);

  v8f acc[4] = {};
#pragma unroll
  for (int kt = 0; kt < 4; ++kt) {
    v16bf a = (kt < 2) ? load_a_bf16(hdp, kt * 32, lane)
                       : load_a_f32(agp, kt * 32 - kD, lane);
#pragma unroll
    for (int nt = 0; nt < 4; ++nt) {
      v16bf b = *(const v16bf*)&wp[((kt * 4 + nt) * 32 + lane) * 16];
      acc[nt] = __builtin_amdgcn_wmma_f32_16x16x32_bf16(
          false, a, false, b, (short)0, acc[nt], false, false);
    }
  }

  const int colb = lane & 15;
  const int half = lane >> 4;
#pragma unroll
  for (int i = 0; i < 8; ++i) {
    int row = m0 + i + half * 8;
    if (row < kN) {
#pragma unroll
      for (int nt = 0; nt < 4; ++nt)
        out[(size_t)row * kD + nt * 16 + colb] = acc[nt][i] + ob[nt];
    }
  }
}

// ---------------------------------------------------------------------------
extern "C" void kernel_launch(void* const* d_in, const int* in_sizes, int n_in,
                              void* d_out, int out_size, void* d_ws,
                              size_t ws_size, hipStream_t stream) {
  (void)in_sizes; (void)n_in; (void)out_size; (void)ws_size;
  const float* feat_src = (const float*)d_in[0];
  const float* feat_dst = (const float*)d_in[1];
  const float* edge_w   = (const float*)d_in[2];
  const int*   src_idx  = (const int*)d_in[3];
  const int*   dst_idx  = (const int*)d_in[4];
  const float* weight_n = (const float*)d_in[5];
  const float* weight_e = (const float*)d_in[6];
  const float* query    = (const float*)d_in[7];
  const float* key_w    = (const float*)d_in[8];
  const float* att_w    = (const float*)d_in[9];
  const float* att_b    = (const float*)d_in[10];
  const float* out_w    = (const float*)d_in[11];
  const float* out_b    = (const float*)d_in[12];
  float* out = (float*)d_out;

  // workspace carve-up (~36 MB)
  char* ws = (char*)d_ws;
  size_t off = 0;
  float*    h_src = (float*)(ws + off);    off += (size_t)kN * kD * sizeof(float);
  float*    agg   = (float*)(ws + off);    off += (size_t)kN * kD * sizeof(float);
  float*    s_q   = (float*)(ws + off);    off += (size_t)kN * sizeof(float);
  float*    s_k   = (float*)(ws + off);    off += (size_t)kN * sizeof(float);
  unsigned* m_enc = (unsigned*)(ws + off); off += (size_t)kN * sizeof(unsigned);
  float*    ssum  = (float*)(ws + off);    off += (size_t)kN * sizeof(float);
  __bf16*   h_dst = (__bf16*)(ws + off);   off += (size_t)kN * kD * sizeof(__bf16);
  float*    exbuf = (float*)(ws + off);    off += (size_t)kE * sizeof(float);
  float*    cst   = (float*)(ws + off);    off += 256 * sizeof(float);

  precompute_kernel<<<1, 64, 0, stream>>>(query, key_w, weight_e, att_w, cst);
  init_kernel<<<(kN * kD + 255) / 256, 256, 0, stream>>>(agg, ssum, m_enc);
  node_transform_kernel<<<(kN + 127) / 128, 256, 0, stream>>>(
      feat_src, feat_dst, weight_n, cst, h_src, h_dst, s_q, s_k);
  edge_logits_kernel<<<1024, 256, 0, stream>>>(edge_w, src_idx, dst_idx, s_q,
                                               s_k, cst, att_b, exbuf, m_enc);
  edge_exp_kernel<<<1024, 256, 0, stream>>>(dst_idx, m_enc, exbuf, ssum);
  edge_scatter_kernel<<<512, 256, 0, stream>>>(src_idx, dst_idx, exbuf, ssum,
                                               h_src, agg);
  out_gemm_kernel<<<(kN + 127) / 128, 256, 0, stream>>>(h_dst, agg, out_w,
                                                        out_b, out);
}
